// PointNetSetAbstraction_36782099923383
// MI455X (gfx1250) — compile-verified, matrix operand order checked by reading, not run on hardware
//
#include <hip/hip_runtime.h>
#include <hip/hip_bf16.h>

typedef __attribute__((ext_vector_type(16))) _Float16 v16h;
typedef __attribute__((ext_vector_type(8)))  _Float16 v8h;
typedef __attribute__((ext_vector_type(8)))  float    v8f;

#define N_PTS 8192
#define S_CEN 2048
#define K_NEI 32
#define BATCH 4
#define D_IN  64

// ---------------------------------------------------------------------------
// FPS: one 1024-thread workgroup per batch. Point cloud lives in LDS (96KB),
// running min-distance lives in registers (8/thread). Each scan step:
// local update + wave32 shuffle argmax + cross-wave LDS argmax.
// ---------------------------------------------------------------------------
__global__ __launch_bounds__(1024) void fps_kernel(const float* __restrict__ xyz,
                                                   float* __restrict__ new_xyz) {
  __shared__ float px[N_PTS], py[N_PTS], pz[N_PTS];
  __shared__ float redV[32];
  __shared__ int   redI[32];
  __shared__ int   curFar;
  const int b = blockIdx.x;
  const int t = threadIdx.x;
  const int lane = t & 31, wid = t >> 5;
  const float* base = xyz + (size_t)b * N_PTS * 3;
  float dreg[8];
  for (int j = 0; j < 8; ++j) {
    int n = t + 1024 * j;
    px[n] = base[3*n+0]; py[n] = base[3*n+1]; pz[n] = base[3*n+2];
    dreg[j] = 1e10f;
  }
  if (t == 0) curFar = 0;
  __syncthreads();
  for (int s = 0; s < S_CEN; ++s) {
    int far = curFar;
    float cx = px[far], cy = py[far], cz = pz[far];
    if (t == 0) {
      float* o = new_xyz + ((size_t)b * S_CEN + s) * 3;
      o[0] = cx; o[1] = cy; o[2] = cz;
    }
    float bV = -1.0f; int bI = 0;
    for (int j = 0; j < 8; ++j) {
      int n = t + 1024 * j;
      float dx = px[n]-cx, dy = py[n]-cy, dz = pz[n]-cz;
      float d = dx*dx + dy*dy + dz*dz;
      float dm = fminf(dreg[j], d);
      dreg[j] = dm;
      if (dm > bV) { bV = dm; bI = n; }
    }
    for (int off = 16; off; off >>= 1) {
      float ov = __shfl_down(bV, off, 32);
      int   oi = __shfl_down(bI, off, 32);
      if (ov > bV) { bV = ov; bI = oi; }
    }
    if (lane == 0) { redV[wid] = bV; redI[wid] = bI; }
    __syncthreads();
    if (wid == 0) {
      bV = redV[lane]; bI = redI[lane];
      for (int off = 16; off; off >>= 1) {
        float ov = __shfl_down(bV, off, 32);
        int   oi = __shfl_down(bI, off, 32);
        if (ov > bV) { bV = ov; bI = oi; }
      }
      if (lane == 0) curFar = bI;
    }
    __syncthreads();
  }
}

// ---------------------------------------------------------------------------
// KNN: point cloud staged in LDS once per block (4 waves, 4 queries/wave).
// Per-lane sorted top-32 candidate lists in LDS (pad 33 -> conflict-free),
// then a 32-round wave argmin merge emits the global 32 nearest.
// ---------------------------------------------------------------------------
#define KNN_WAVES 4
#define KNN_QPW   4
__global__ __launch_bounds__(128) void knn_kernel(const float* __restrict__ xyz,
                                                  const float* __restrict__ new_xyz,
                                                  int* __restrict__ knn_idx) {
  __shared__ float px[N_PTS], py[N_PTS], pz[N_PTS];
  __shared__ float lstD[KNN_WAVES][32][33];
  __shared__ int   lstI[KNN_WAVES][32][33];
  const int b = blockIdx.y;
  const int t = threadIdx.x;
  const int lane = t & 31, wid = t >> 5;
  const float* base = xyz + (size_t)b * N_PTS * 3;
  for (int i = t; i < N_PTS; i += 128) {
    px[i] = base[3*i+0]; py[i] = base[3*i+1]; pz[i] = base[3*i+2];
  }
  __syncthreads();
  for (int qq = 0; qq < KNN_QPW; ++qq) {
    int s = (blockIdx.x * KNN_WAVES + wid) * KNN_QPW + qq;
    const float* q = new_xyz + ((size_t)b * S_CEN + s) * 3;
    float qx = q[0], qy = q[1], qz = q[2];
    float* Ld = lstD[wid][lane];
    int*   Li = lstI[wid][lane];
    int cnt = 0;
    float worst = 1e30f;
    for (int c = 0; c < N_PTS/32; ++c) {
      int n = lane + 32 * c;
      float dx = px[n]-qx, dy = py[n]-qy, dz = pz[n]-qz;
      float d = dx*dx + dy*dy + dz*dz;
      if (d < worst || cnt < K_NEI) {
        int pos = (cnt < K_NEI) ? cnt : (K_NEI - 1);
        while (pos > 0 && Ld[pos-1] > d) {
          Ld[pos] = Ld[pos-1]; Li[pos] = Li[pos-1]; --pos;
        }
        Ld[pos] = d; Li[pos] = n;
        if (cnt < K_NEI) ++cnt;
        worst = (cnt == K_NEI) ? Ld[K_NEI-1] : 1e30f;
      }
    }
    int ptr = 0;
    for (int r = 0; r < K_NEI; ++r) {
      float hv = (ptr < K_NEI) ? Ld[ptr] : 1e30f;
      int hidx  = (ptr < K_NEI) ? Li[ptr] : 0;
      float bv = hv; int bl = lane;
      for (int off = 16; off; off >>= 1) {
        float ov = __shfl_xor(bv, off, 32);
        int   ol = __shfl_xor(bl, off, 32);
        if (ov < bv || (ov == bv && ol < bl)) { bv = ov; bl = ol; }
      }
      int widx = __shfl(hidx, bl, 32);
      if (lane == bl) ++ptr;
      if (lane == 0) knn_idx[((size_t)b * S_CEN + s) * K_NEI + r] = widx;
    }
  }
}

// ---------------------------------------------------------------------------
// Gather + 3-layer MLP on v_wmma_f32_16x16x32_f16 + K-maxpool.
// One wave per centroid (M=32 neighbors -> 2 M-tiles). Weights staged in LDS
// N-major so B-fragments are two contiguous ds_load_b128 per lane, matching
// the ISA 16-bit fragment layout: elems 0-7 <-> K=(0..7)+8*hi, elems 8-15
// <-> K=16+(0..7)+8*hi (hi = lane>=16). BN folded into scale/bias on the
// fp32 accumulators; layer-2 tiles are max-pooled in registers (D layout:
// lane = column, reg r = row m0+r+8*hi) and halves merged via shfl_xor(16).
// ---------------------------------------------------------------------------
__device__ __forceinline__ v16h ldfrag(const _Float16* p) {
  union { v16h v; v8h h[2]; } u;
  u.h[0] = *(const v8h*)(p);
  u.h[1] = *(const v8h*)(p + 16);
  return u.v;
}

#define MLP_WAVES 8
__global__ __launch_bounds__(256) void group_mlp_kernel(
    const float* __restrict__ xyz, const float* __restrict__ points,
    const int* __restrict__ knn_idx, const float* __restrict__ new_xyz,
    const float* __restrict__ w0, const float* __restrict__ g0, const float* __restrict__ b0,
    const float* __restrict__ m0, const float* __restrict__ v0,
    const float* __restrict__ w1, const float* __restrict__ g1, const float* __restrict__ b1,
    const float* __restrict__ m1, const float* __restrict__ v1,
    const float* __restrict__ w2, const float* __restrict__ g2, const float* __restrict__ b2,
    const float* __restrict__ m2, const float* __restrict__ v2,
    float* __restrict__ out_pts) {
  __shared__ __align__(16) _Float16 w0N[64][96];    // W0 row-major (n,k), K padded 67->96
  __shared__ __align__(16) _Float16 w1N[64][64];
  __shared__ __align__(16) _Float16 w2N[128][64];
  __shared__ float sc0[64], bi0[64], sc1[64], bi1[64], sc2[128], bi2[128];
  __shared__ __align__(16) _Float16 actA[MLP_WAVES][32][96];  // L0 input / L1 output (as [32][64])
  __shared__ __align__(16) _Float16 actB[MLP_WAVES][32][64];  // L0 output / L1 input

  const int t = threadIdx.x;
  for (int i = t; i < 64*96; i += 256) {
    int n = i / 96, k = i - n * 96;
    w0N[n][k] = (k < 67) ? (_Float16)w0[n*67 + k] : (_Float16)0.0f;
  }
  for (int i = t; i < 64*64;  i += 256) w1N[i>>6][i&63] = (_Float16)w1[i];
  for (int i = t; i < 128*64; i += 256) w2N[i>>6][i&63] = (_Float16)w2[i];
  for (int i = t; i < 64; i += 256) {
    float s0 = g0[i] * rsqrtf(v0[i] + 1e-5f); sc0[i] = s0; bi0[i] = b0[i] - m0[i]*s0;
    float s1 = g1[i] * rsqrtf(v1[i] + 1e-5f); sc1[i] = s1; bi1[i] = b1[i] - m1[i]*s1;
  }
  for (int i = t; i < 128; i += 256) {
    float s2 = g2[i] * rsqrtf(v2[i] + 1e-5f); sc2[i] = s2; bi2[i] = b2[i] - m2[i]*s2;
  }
  __syncthreads();

  const int lane = t & 31, wid = t >> 5;
  const int qi = blockIdx.x * MLP_WAVES + wid;   // = b*S + s
  const int b = qi >> 11, s = qi & 2047;
  const int hi = lane >> 4;
  const int lo = lane & 15;

  // ---- gather: lane = neighbor; build A0 = [32 x 96] f16 (rel xyz | feats | 0pad)
  {
    int n = knn_idx[(size_t)qi * K_NEI + lane];
    const float* q = new_xyz + (size_t)qi * 3;
    const float* p = xyz + ((size_t)b * N_PTS + n) * 3;
    _Float16* row = actA[wid][lane];
    row[0] = (_Float16)(p[0] - q[0]);
    row[1] = (_Float16)(p[1] - q[1]);
    row[2] = (_Float16)(p[2] - q[2]);
    const float* f = points + (size_t)b * D_IN * N_PTS + n;
    #pragma unroll 8
    for (int d = 0; d < D_IN; ++d) row[3 + d] = (_Float16)f[(size_t)d * N_PTS];
    for (int k = 67; k < 96; ++k) row[k] = (_Float16)0.0f;
  }

  // ---- layer 0: (32x96)x(96x64) ----
  {
    const _Float16* A = &actA[wid][0][0];
    for (int nt = 0; nt < 4; ++nt) {
      int col = nt*16 + lo;
      float s_ = sc0[col], o_ = bi0[col];
      for (int mt = 0; mt < 2; ++mt) {
        v8f acc = {0.f,0.f,0.f,0.f,0.f,0.f,0.f,0.f};
        for (int kc = 0; kc < 3; ++kc) {
          v16h a = ldfrag(A + (mt*16 + lo)*96 + kc*32 + 8*hi);
          v16h w = ldfrag(&w0N[0][0] + col*96 + kc*32 + 8*hi);
          acc = __builtin_amdgcn_wmma_f32_16x16x32_f16(false, a, false, w, (short)0, acc, false, false);
        }
        int rb = mt*16 + 8*hi;
        for (int r = 0; r < 8; ++r)
          actB[wid][rb + r][col] = (_Float16)fmaxf(acc[r]*s_ + o_, 0.0f);
      }
    }
  }

  // ---- layer 1: (32x64)x(64x64) ----
  {
    const _Float16* A = &actB[wid][0][0];
    _Float16* O = &actA[wid][0][0];       // reuse as [32][64]
    for (int nt = 0; nt < 4; ++nt) {
      int col = nt*16 + lo;
      float s_ = sc1[col], o_ = bi1[col];
      for (int mt = 0; mt < 2; ++mt) {
        v8f acc = {0.f,0.f,0.f,0.f,0.f,0.f,0.f,0.f};
        for (int kc = 0; kc < 2; ++kc) {
          v16h a = ldfrag(A + (mt*16 + lo)*64 + kc*32 + 8*hi);
          v16h w = ldfrag(&w1N[0][0] + col*64 + kc*32 + 8*hi);
          acc = __builtin_amdgcn_wmma_f32_16x16x32_f16(false, a, false, w, (short)0, acc, false, false);
        }
        int rb = mt*16 + 8*hi;
        for (int r = 0; r < 8; ++r)
          O[(rb + r)*64 + col] = (_Float16)fmaxf(acc[r]*s_ + o_, 0.0f);
      }
    }
  }

  // ---- layer 2: (32x64)x(64x128) + maxpool over the 32 neighbors ----
  {
    const _Float16* A = &actA[wid][0][0]; // [32][64]
    for (int nt = 0; nt < 8; ++nt) {
      int col = nt*16 + lo;
      float s_ = sc2[col], o_ = bi2[col];
      float cmax = 0.0f;                  // ReLU outputs are >= 0
      for (int mt = 0; mt < 2; ++mt) {
        v8f acc = {0.f,0.f,0.f,0.f,0.f,0.f,0.f,0.f};
        for (int kc = 0; kc < 2; ++kc) {
          v16h a = ldfrag(A + (mt*16 + lo)*64 + kc*32 + 8*hi);
          v16h w = ldfrag(&w2N[0][0] + col*64 + kc*32 + 8*hi);
          acc = __builtin_amdgcn_wmma_f32_16x16x32_f16(false, a, false, w, (short)0, acc, false, false);
        }
        for (int r = 0; r < 8; ++r)
          cmax = fmaxf(cmax, fmaxf(acc[r]*s_ + o_, 0.0f));
      }
      cmax = fmaxf(cmax, __shfl_xor(cmax, 16, 32));   // merge hi/lo half-wave rows
      if (lane < 16)
        out_pts[((size_t)b * 128 + col) * S_CEN + s] = cmax;
    }
  }
}

// ---------------------------------------------------------------------------
extern "C" void kernel_launch(void* const* d_in, const int* in_sizes, int n_in,
                              void* d_out, int out_size, void* d_ws, size_t ws_size,
                              hipStream_t stream) {
  const float* xyz    = (const float*)d_in[0];
  const float* points = (const float*)d_in[1];
  const float* w0 = (const float*)d_in[2];
  const float* g0 = (const float*)d_in[3];
  const float* b0 = (const float*)d_in[4];
  const float* m0 = (const float*)d_in[5];
  const float* v0 = (const float*)d_in[6];
  const float* w1 = (const float*)d_in[7];
  const float* g1 = (const float*)d_in[8];
  const float* b1 = (const float*)d_in[9];
  const float* m1 = (const float*)d_in[10];
  const float* v1 = (const float*)d_in[11];
  const float* w2 = (const float*)d_in[12];
  const float* g2 = (const float*)d_in[13];
  const float* b2 = (const float*)d_in[14];
  const float* m2 = (const float*)d_in[15];
  const float* v2 = (const float*)d_in[16];

  float* new_xyz = (float*)d_out;                                  // (B,S,3)
  float* out_pts = (float*)d_out + (size_t)BATCH * S_CEN * 3;      // (B,128,S)
  int*   knn_idx = (int*)d_ws;                                     // (B,S,K)

  fps_kernel<<<BATCH, 1024, 0, stream>>>(xyz, new_xyz);
  knn_kernel<<<dim3(S_CEN / (KNN_WAVES * KNN_QPW), BATCH), 128, 0, stream>>>(xyz, new_xyz, knn_idx);
  group_mlp_kernel<<<(BATCH * S_CEN) / MLP_WAVES, 256, 0, stream>>>(
      xyz, points, knn_idx, new_xyz,
      w0, g0, b0, m0, v0, w1, g1, b1, m1, v1, w2, g2, b2, m2, v2, out_pts);
}
